// MultiHeadAttention_31250182045832
// MI455X (gfx1250) — matrix operand order repaired and checked by reading, not compiled, not run on hardware
//
#include <hip/hip_runtime.h>
#include <hip/hip_bf16.h>
#include <stdint.h>

#define B_ 8
#define S_ 1024
#define E_ 1024
#define H_ 16
#define D_ 64

typedef __attribute__((ext_vector_type(16))) __bf16 bf16x16;
typedef __attribute__((ext_vector_type(8)))  float  floatx8;
typedef unsigned int u32x4 __attribute__((ext_vector_type(4)));
typedef int          i32x4 __attribute__((ext_vector_type(4)));
typedef int          i32x8 __attribute__((ext_vector_type(8)));

#if defined(__has_builtin)
#if __has_builtin(__builtin_amdgcn_tensor_load_to_lds)
#define HAVE_TDM 1
#endif
#endif
#ifndef HAVE_TDM
#define HAVE_TDM 0
#endif

union FragAB { uint4 q[2]; bf16x16 v; };

__device__ __forceinline__ unsigned short f2bf(float f) {
  unsigned int u = __builtin_bit_cast(unsigned int, f);
  u += 0x7FFFu + ((u >> 16) & 1u);           // round-to-nearest-even
  return (unsigned short)(u >> 16);
}

// A fragment: 16x32 bf16, lanes 0-15 = rows (K octets 0-7,16-23), lanes 16-31 same rows (K 8-15,24-31)
__device__ __forceinline__ bf16x16 load_a_frag(const unsigned short* base, int stride,
                                               int row0, int k0, int lane) {
  const int grp = lane >> 4;
  const int m   = lane & 15;
  FragAB f;
  const unsigned short* p = base + (size_t)(row0 + m) * stride + k0 + grp * 8;
  f.q[0] = *(const uint4*)p;          // K octet 0 (or 8)
  f.q[1] = *(const uint4*)(p + 16);   // K octet 16 (or 24)
  return f.v;
}

// B fragment: 32x16 bf16 (KxN), lane%16 = N, elements = K (octets 0-7,16-23 / 8-15,24-31)
__device__ __forceinline__ bf16x16 load_b_frag(const unsigned short* base, int stride,
                                               int k0, int n0, int lane) {
  FragAB f;
  const int grp = lane >> 4;
  const unsigned short* p = base + (size_t)k0 * stride + n0 + (lane & 15);
#pragma unroll
  for (int s = 0; s < 16; ++s) {
    const int k = (s < 8) ? (grp * 8 + s) : (16 + grp * 8 + (s - 8));
    f.v[s] = __builtin_bit_cast(__bf16, p[(size_t)k * stride]);
  }
  return f.v;
}

#if HAVE_TDM
// TDM: DMA a [tile_h x 64] bf16 tile (row stride 64 elems in memory) into LDS,
// padding 16B after each 128B row so the LDS row stride is 144B (= 72 ushorts = TSTR).
// tensor dims == tile dims, so no OOB path is exercised.
__device__ __forceinline__ void tdm_load_bf16_tile(unsigned int lds_off,
                                                   const unsigned short* g,
                                                   int tile_w, int tile_h) {
  unsigned long long ga = (unsigned long long)(size_t)g;
  u32x4 g0 = {0u, 0u, 0u, 0u};
  g0[0] = 1u;                                            // count=1, user descriptor
  g0[1] = lds_off;                                       // lds_addr (bytes)
  g0[2] = (unsigned int)ga;                              // global_addr[31:0]
  g0[3] = (unsigned int)((ga >> 32) & 0x01FFFFFFull)     // global_addr[56:32]
        | 0x80000000u;                                   // type=2 ("image")
  i32x8 g1 = {0, 0, 0, 0, 0, 0, 0, 0};
  g1[0] = 0x00010000                                     // data_size=1 (2 bytes)
        | (1 << 20)                                      // pad_enable
        | (4 << 22)                                      // pad_interval: 32 DWORDs (128B)
        | (3 << 25);                                     // pad_amount: 4 DWORDs (16B)
  g1[1] = (tile_w & 0xFFFF) << 16;                       // tensor_dim0[15:0]
  g1[2] = (tile_h & 0xFFFF) << 16;                       // tensor_dim0 hi=0, tensor_dim1[15:0]
  g1[3] = (tile_w & 0xFFFF) << 16;                       // tensor_dim1 hi=0, tile_dim0
  g1[4] = (tile_h & 0xFFFF);                             // tile_dim1 (tile_dim2=0)
  g1[5] = tile_w;                                        // tensor_dim0_stride[31:0]
  i32x4 gz = {0, 0, 0, 0};
#if __clang_major__ >= 23
  i32x8 gz8 = {0, 0, 0, 0, 0, 0, 0, 0};
  __builtin_amdgcn_tensor_load_to_lds(g0, g1, gz, gz, gz8, 0);
#else
  __builtin_amdgcn_tensor_load_to_lds(g0, g1, gz, gz, 0);
#endif
}
#endif

// ---------------------------------------------------------------------------
// Kernel 1: y = X @ W^T + b, fp32 in, bf16 out laid out [B,H,S,D]
// ---------------------------------------------------------------------------
#define PBM 64
#define PBN 64
#define PBK 32
#define ASTR 40
#define BSTR 72

__global__ __launch_bounds__(256)
void qkv_proj_kernel(const float* __restrict__ X, const float* __restrict__ W,
                     const float* __restrict__ bias, unsigned short* __restrict__ out) {
  __shared__ __align__(16) unsigned short As[PBM][ASTR];   // [m][k]
  __shared__ __align__(16) unsigned short Bs[PBK][BSTR];   // [k][n]

  const int tid  = threadIdx.x;
  const int lane = tid & 31;
  const int wave = tid >> 5;          // 0..7
  const int wm   = wave & 3;          // 4 waves over M
  const int wn   = wave >> 2;         // 2 waves over N
  const int mBase = blockIdx.y * PBM;
  const int nBase = blockIdx.x * PBN;

  const floatx8 z = {0.f,0.f,0.f,0.f,0.f,0.f,0.f,0.f};
  floatx8 acc[2] = {z, z};

  const int lrow = tid >> 2;          // 0..63
  const int lkq  = (tid & 3) * 8;     // 0,8,16,24

  for (int k0 = 0; k0 < E_; k0 += PBK) {
    // stage A tile: X[mBase+row][k0..k0+31]  fp32 -> bf16
    {
      const float* px = X + (size_t)(mBase + lrow) * E_ + k0 + lkq;
      float4 a0 = *(const float4*)px;
      float4 a1 = *(const float4*)(px + 4);
      uint4 pk;
      pk.x = (unsigned int)f2bf(a0.x) | ((unsigned int)f2bf(a0.y) << 16);
      pk.y = (unsigned int)f2bf(a0.z) | ((unsigned int)f2bf(a0.w) << 16);
      pk.z = (unsigned int)f2bf(a1.x) | ((unsigned int)f2bf(a1.y) << 16);
      pk.w = (unsigned int)f2bf(a1.z) | ((unsigned int)f2bf(a1.w) << 16);
      *(uint4*)&As[lrow][lkq] = pk;
    }
    // stage B tile transposed: Bs[k][n] = W[nBase+n][k0+k]
    {
      const float* pw = W + (size_t)(nBase + lrow) * E_ + k0 + lkq;
      float4 b0 = *(const float4*)pw;
      float4 b1 = *(const float4*)(pw + 4);
      float vb[8] = {b0.x,b0.y,b0.z,b0.w,b1.x,b1.y,b1.z,b1.w};
#pragma unroll
      for (int i = 0; i < 8; ++i) Bs[lkq + i][lrow] = f2bf(vb[i]);
    }
    __syncthreads();

    bf16x16 afr = load_a_frag(&As[0][0], ASTR, wm * 16, 0, lane);
#pragma unroll
    for (int tn = 0; tn < 2; ++tn) {
      bf16x16 bfr = load_b_frag(&Bs[0][0], BSTR, 0, wn * 32 + tn * 16, lane);
      acc[tn] = __builtin_amdgcn_wmma_f32_16x16x32_bf16(
          false, afr, false, bfr, (short)0, acc[tn], false, false);
    }
    __syncthreads();
  }

  // epilogue: + bias, convert bf16, scatter to [B,H,S,D]
  const int grp = lane >> 4;
  const int nn  = lane & 15;
#pragma unroll
  for (int tn = 0; tn < 2; ++tn) {
    const int ncol = nBase + wn * 32 + tn * 16 + nn;
    const float bv = bias[ncol];
    const int h = ncol >> 6, d = ncol & 63;
#pragma unroll
    for (int r = 0; r < 8; ++r) {
      const int m = mBase + wm * 16 + grp * 8 + r;
      const int b = m >> 10, s = m & 1023;
      out[(((size_t)(b * H_ + h)) * S_ + s) * D_ + d] = f2bf(acc[tn][r] + bv);
    }
  }
}

// ---------------------------------------------------------------------------
// Kernel 2: per (b,h): O = softmax(tril_as_zeros(QK^T/8)) V   (streaming)
// ---------------------------------------------------------------------------
#define QB 64
#define KB 64
#define TSTR 72
#define SSTR 66

__global__ __launch_bounds__(256)
void attn_kernel(const unsigned short* __restrict__ Qw, const unsigned short* __restrict__ Kw,
                 const unsigned short* __restrict__ Vw, float* __restrict__ Out) {
  __shared__ __align__(16) unsigned short Qs [QB][TSTR];   // [i][d]
  __shared__ __align__(16) unsigned short KTs[D_][TSTR];   // [d][j]  (K transposed)
  __shared__ __align__(16) unsigned short Vs [KB][TSTR];   // [j][d]
  __shared__ __align__(16) unsigned short Ps [QB][TSTR];   // [i][j]  (probs, bf16)
  __shared__ float Ss[QB][SSTR];                            // raw scaled scores
  __shared__ __align__(32) float mRow[QB];
  __shared__ __align__(32) float lRow[QB];
  __shared__ __align__(32) float scRow[QB];
  __shared__ float red[QB][4];

  const int tid  = threadIdx.x;
  const int lane = tid & 31;
  const int wave = tid >> 5;
  const int wm   = wave & 3;           // 4 waves over query rows
  const int wn   = wave >> 2;          // 2 waves over cols (j or d)
  const int bh   = blockIdx.y;         // 0..127
  const int b    = bh >> 4, h = bh & 15;
  const int qBase = blockIdx.x * QB;

  const unsigned short* Qh = Qw + (size_t)bh * S_ * D_;
  const unsigned short* Kh = Kw + (size_t)bh * S_ * D_;
  const unsigned short* Vh = Vw + (size_t)bh * S_ * D_;

  // load Q block once: 64 rows x 64 bf16
#if HAVE_TDM
  if (wave == 0) {
    tdm_load_bf16_tile((unsigned int)(size_t)&Qs[0][0], Qh + (size_t)qBase * D_, D_, QB);
    __builtin_amdgcn_s_wait_tensorcnt(0);
  }
#else
  {
    int idx = tid;
#pragma unroll
    for (int r = 0; r < 2; ++r, idx += 256) {
      const int row = idx >> 3;
      const int c8  = (idx & 7) * 8;
      *(uint4*)&Qs[row][c8] = *(const uint4*)(Qh + (size_t)(qBase + row) * D_ + c8);
    }
  }
#endif
  if (tid < QB) { mRow[tid] = -1e30f; lRow[tid] = 0.f; }

  const floatx8 z = {0.f,0.f,0.f,0.f,0.f,0.f,0.f,0.f};
  floatx8 accO[2] = {z, z};
  __syncthreads();

  const int ldJ = tid >> 2;            // 0..63  (row for loads & softmax)
  const int ldQ = (tid & 3) * 16;      // 16-wide column segment

  for (int t = 0; t < 16; ++t) {
    const int jBase = t * KB;
#if HAVE_TDM
    if (wave == 0)   // async DMA of the V tile overlaps the K transpose below
      tdm_load_bf16_tile((unsigned int)(size_t)&Vs[0][0], Vh + (size_t)jBase * D_, D_, KB);
#endif
    // stage K^T tile (transpose d<->j; TDM cannot transpose)
    {
      const unsigned short* pk = Kh + (size_t)(jBase + ldJ) * D_ + ldQ;
      uint4 k0v = *(const uint4*)pk;
      uint4 k1v = *(const uint4*)(pk + 8);
      const unsigned int w[8] = {k0v.x,k0v.y,k0v.z,k0v.w,k1v.x,k1v.y,k1v.z,k1v.w};
#pragma unroll
      for (int i = 0; i < 8; ++i) {
        KTs[ldQ + 2*i + 0][ldJ] = (unsigned short)(w[i] & 0xFFFFu);
        KTs[ldQ + 2*i + 1][ldJ] = (unsigned short)(w[i] >> 16);
      }
#if !HAVE_TDM
      const unsigned short* pv = Vh + (size_t)(jBase + ldJ) * D_ + ldQ;
      *(uint4*)&Vs[ldJ][ldQ]     = *(const uint4*)pv;
      *(uint4*)&Vs[ldJ][ldQ + 8] = *(const uint4*)(pv + 8);
#endif
      if (t < 15)
        __builtin_prefetch(Kh + (size_t)(jBase + KB + ldJ) * D_ + ldQ, 0, 1);
    }
#if HAVE_TDM
    if (wave == 0) __builtin_amdgcn_s_wait_tensorcnt(0);
#endif
    __syncthreads();

    // S = Q K^T / sqrt(D)
#pragma unroll
    for (int tn = 0; tn < 2; ++tn) {
      floatx8 sAcc = z;
#pragma unroll
      for (int ks = 0; ks < D_; ks += 32) {
        bf16x16 afr = load_a_frag(&Qs[0][0],  TSTR, wm * 16, ks, lane);
        bf16x16 bfr = load_b_frag(&KTs[0][0], TSTR, ks, wn * 32 + tn * 16, lane);
        sAcc = __builtin_amdgcn_wmma_f32_16x16x32_bf16(
            false, afr, false, bfr, (short)0, sAcc, false, false);
      }
      const int jj = wn * 32 + tn * 16 + (lane & 15);
#pragma unroll
      for (int r = 0; r < 8; ++r) {
        const int ii = wm * 16 + (lane >> 4) * 8 + r;
        Ss[ii][jj] = sAcc[r] * 0.125f;
      }
    }
    __syncthreads();

    // streaming softmax with the reference quirk: masked entries are 0.0, not -inf
    {
      const int gi = qBase + ldJ;
      const float m_old = mRow[ldJ];
      float vloc[16];
      float mx = -1e30f;
#pragma unroll
      for (int c = 0; c < 16; ++c) {
        const int gj = jBase + ldQ + c;
        const float v = (gj > gi) ? 0.f : Ss[ldJ][ldQ + c];
        vloc[c] = v;
        mx = fmaxf(mx, v);
      }
      red[ldJ][tid & 3] = mx;
      __syncthreads();
      const float m_blk = fmaxf(fmaxf(red[ldJ][0], red[ldJ][1]),
                                fmaxf(red[ldJ][2], red[ldJ][3]));
      const float m_new = fmaxf(m_old, m_blk);
      __syncthreads();
      float sum = 0.f;
#pragma unroll
      for (int c = 0; c < 16; ++c) {
        const float p = __expf(vloc[c] - m_new);
        sum += p;
        Ps[ldJ][ldQ + c] = f2bf(p);
      }
      red[ldJ][tid & 3] = sum;
      __syncthreads();
      if ((tid & 3) == 0) {
        const float bsum = red[ldJ][0] + red[ldJ][1] + red[ldJ][2] + red[ldJ][3];
        const float scale = __expf(m_old - m_new);
        lRow[ldJ]  = lRow[ldJ] * scale + bsum;
        mRow[ldJ]  = m_new;
        scRow[ldJ] = scale;
      }
      __syncthreads();
    }

    // O = O*scale + P V
    {
      const floatx8 sc = *(const floatx8*)&scRow[wm * 16 + (lane >> 4) * 8];
      accO[0] *= sc;
      accO[1] *= sc;
#pragma unroll
      for (int ks = 0; ks < KB; ks += 32) {
        bf16x16 afr = load_a_frag(&Ps[0][0], TSTR, wm * 16, ks, lane);
#pragma unroll
        for (int tn = 0; tn < 2; ++tn) {
          bf16x16 bfr = load_b_frag(&Vs[0][0], TSTR, ks, wn * 32 + tn * 16, lane);
          accO[tn] = __builtin_amdgcn_wmma_f32_16x16x32_bf16(
              false, afr, false, bfr, (short)0, accO[tn], false, false);
        }
      }
    }
    __syncthreads();
  }

  // epilogue: divide by l, store fp32 at flat [B,S,H,D] (== reshape(-1,B,E) view)
  {
    const floatx8 lv = *(const floatx8*)&lRow[wm * 16 + (lane >> 4) * 8];
#pragma unroll
    for (int tn = 0; tn < 2; ++tn) {
      const int d = wn * 32 + tn * 16 + (lane & 15);
#pragma unroll
      for (int r = 0; r < 8; ++r) {
        const int i = qBase + wm * 16 + (lane >> 4) * 8 + r;
        Out[((size_t)(b * S_ + i)) * E_ + h * D_ + d] = accO[tn][r] / lv[r];
      }
    }
  }
}

// ---------------------------------------------------------------------------
extern "C" void kernel_launch(void* const* d_in, const int* in_sizes, int n_in,
                              void* d_out, int out_size, void* d_ws, size_t ws_size,
                              hipStream_t stream) {
  const float* queries = (const float*)d_in[0];
  const float* keys    = (const float*)d_in[1];
  const float* values  = (const float*)d_in[2];
  const float* Wq = (const float*)d_in[3];
  const float* bq = (const float*)d_in[4];
  const float* Wk = (const float*)d_in[5];
  const float* bk = (const float*)d_in[6];
  const float* Wv = (const float*)d_in[7];
  const float* bv = (const float*)d_in[8];

  unsigned short* qws = (unsigned short*)d_ws;                       // [B,H,S,D] bf16
  unsigned short* kws = qws + (size_t)B_ * S_ * E_;
  unsigned short* vws = kws + (size_t)B_ * S_ * E_;

  dim3 pg(E_ / PBN, (B_ * S_) / PBM);   // (16, 128)
  qkv_proj_kernel<<<pg, 256, 0, stream>>>(queries, Wq, bq, qws);
  qkv_proj_kernel<<<pg, 256, 0, stream>>>(keys,    Wk, bk, kws);
  qkv_proj_kernel<<<pg, 256, 0, stream>>>(values,  Wv, bv, vws);

  dim3 ag(S_ / QB, B_ * H_);            // (16, 128)
  attn_kernel<<<ag, 256, 0, stream>>>(qws, kws, vws, (float*)d_out);
}
